// GNNLayer_34840774705589
// MI455X (gfx1250) — matrix operand order repaired and checked by reading, not compiled, not run on hardware
//
#include <hip/hip_runtime.h>
#include <hip/hip_bf16.h>
#include <math.h>

// MI455X (gfx1250) implementation.
// Pipeline:
//   1. zero agg region of d_out (atomic accumulation target)
//   2. node flags (trivial)
//   3. RW = rela_embed @ Wr ; QW = rela_embed[q_rel] @ Wqr_w + b  (tiny)
//   4. HsW = hidden @ Ws           -- bf16 WMMA GEMM (N x 128 x 128)
//   5. per-edge: pre=relu(HsW[sub]+RW[rel]+QW[ridx]); alpha=sigmoid(pre.walpha);
//      atomicAdd alpha*(hidden[sub]+rela[rel]) into agg[obj]   (wave per edge)
//   6. hidden_new = agg @ Wh       -- same WMMA GEMM, in-place in d_out

typedef __attribute__((ext_vector_type(16))) __bf16 v16bf;
typedef __attribute__((ext_vector_type(8)))  __bf16 v8bf;
typedef __attribute__((ext_vector_type(8)))  float  v8f;

#define DD 128

__global__ __launch_bounds__(256) void zero_f4(float* __restrict__ p, int n4) {
    int i = blockIdx.x * 256 + threadIdx.x;
    if (i < n4) ((float4*)p)[i] = make_float4(0.f, 0.f, 0.f, 0.f);
}

__global__ __launch_bounds__(256) void node_flags(const int* __restrict__ nodes,
                                                  float* __restrict__ flags,
                                                  int N, int n_node) {
    int i = blockIdx.x * blockDim.x + threadIdx.x;
    if (i < N) {
        int v = nodes[2 * i + 1];
        flags[i] = (v > -1 && v < n_node + 1) ? 1.0f : 0.0f;
    }
}

// One block per output row (REL_VOCAB rows of RW, then B rows of QW), 128 threads = cols.
__global__ __launch_bounds__(128) void precompute_rw_qw(
    const float* __restrict__ rela, const int* __restrict__ q_rel,
    const float* __restrict__ Wr, const float* __restrict__ Wqr_w,
    const float* __restrict__ Wqr_b,
    float* __restrict__ RW, float* __restrict__ QW, int rel_vocab) {
    int c = threadIdx.x;
    int row = blockIdx.x;
    if (row < rel_vocab) {
        const float* h = rela + (size_t)row * DD;
        float acc = 0.f;
        for (int k = 0; k < DD; k++) acc += h[k] * Wr[k * DD + c];
        RW[row * DD + c] = acc;
    } else {
        int b = row - rel_vocab;
        const float* h = rela + (size_t)q_rel[b] * DD;
        float acc = Wqr_b[c];
        for (int k = 0; k < DD; k++) acc += h[k] * Wqr_w[k * DD + c];
        QW[b * DD + c] = acc;
    }
}

// Y[nrows x 128] = X[nrows x 128] @ W[128 x 128], bf16 WMMA, f32 accumulate.
// Safe for X == Y: all reads of X are staged to LDS before the barrier;
// all writes happen after it, and a block only writes its own row range.
__global__ __launch_bounds__(256) void gemm_x128_w128(const float* X,
                                                      const float* __restrict__ W,
                                                      float* Y, int nrows) {
    __shared__ __align__(16) __bf16 sX[128 * 128];  // 32 KB
    __shared__ __align__(16) __bf16 sW[128 * 128];  // 32 KB
    const int tid = threadIdx.x;
    const int rowbase = blockIdx.x * 128;

    // Stage W (k-major 128x128) as bf16, coalesced float4 reads.
    for (int i = tid; i < (128 * 128) / 4; i += 256) {
        float4 w = ((const float4*)W)[i];
        int e = i * 4;
        sW[e + 0] = (__bf16)w.x; sW[e + 1] = (__bf16)w.y;
        sW[e + 2] = (__bf16)w.z; sW[e + 3] = (__bf16)w.w;
    }
    // Stage this block's 128 rows of X (guarded; OOB rows -> 0).
    for (int i = tid; i < (128 * 128) / 4; i += 256) {
        int e = i * 4;
        int r = e >> 7;
        float4 x = make_float4(0.f, 0.f, 0.f, 0.f);
        if (rowbase + r < nrows)
            x = ((const float4*)(X + (size_t)(rowbase + r) * 128))[(e & 127) >> 2];
        sX[e + 0] = (__bf16)x.x; sX[e + 1] = (__bf16)x.y;
        sX[e + 2] = (__bf16)x.z; sX[e + 3] = (__bf16)x.w;
    }
    __syncthreads();

    const int wid   = tid >> 5;    // wave id = n-tile (0..7)
    const int lane  = tid & 31;
    const int l16   = lane & 15;
    const int khalf = lane >> 4;   // 0 or 1
    const int col   = wid * 16 + l16;

    union Frag { v16bf v; v8bf h[2]; };

    // B fragments (contiguous k order, matching A: lanes 0-15 hold K 0..15 packed 2/VGPR,
    // lanes 16-31 hold K 16..31; column = lane%16).
    Frag bfrag[4];
#pragma unroll
    for (int kt = 0; kt < 4; kt++) {
#pragma unroll
        for (int j = 0; j < 16; j++)
            bfrag[kt].v[j] = sW[(kt * 32 + khalf * 16 + j) * 128 + col];
    }

#pragma unroll
    for (int rt = 0; rt < 8; rt++) {
        int rb = rowbase + rt * 16;
        if (rb < nrows) {   // nrows is a multiple of 16: whole-tile guard only
            v8f acc = {};
#pragma unroll
            for (int kt = 0; kt < 4; kt++) {
                // A layout: lane<16 row=l16 holds K{kb..kb+15}, kb=kt*32;
                // lane>=16 holds K{kb+16..kb+31}.
                Frag a;
                const __bf16* src = &sX[(rt * 16 + l16) * 128 + kt * 32 + khalf * 16];
                a.h[0] = *(const v8bf*)(src);
                a.h[1] = *(const v8bf*)(src + 8);
                acc = __builtin_amdgcn_wmma_f32_16x16x32_bf16(
                    false, a.v, false, bfrag[kt].v, (short)0, acc, false, false);
            }
            // C/D layout: VGPR j -> M=j (lanes 0-15) / M=j+8 (lanes 16-31), N=lane%16.
            int rowoff = khalf * 8;
#pragma unroll
            for (int j = 0; j < 8; j++)
                Y[(size_t)(rb + rowoff + j) * 128 + col] = acc[j];
        }
    }
}

// One wave (32 lanes, wave32) per edge; lane owns 4 consecutive channels.
__global__ __launch_bounds__(256) void edge_kernel(
    const int* __restrict__ edges, const float* __restrict__ hidden,
    const float* __restrict__ rela, const float* __restrict__ HsW,
    const float* __restrict__ RW, const float* __restrict__ QW,
    const float* __restrict__ walpha_w, const float* __restrict__ walpha_b,
    float* __restrict__ agg, float* __restrict__ alpha_out, int E) {
    int gid  = blockIdx.x * 8 + (threadIdx.x >> 5);
    int lane = threadIdx.x & 31;
    if (gid >= E) return;

    const int* e = edges + (size_t)gid * 6;
    int ridx = e[0], rel = e[2], sub = e[4], obj = e[5];
    int c = lane * 4;

    float4 hs = *(const float4*)(HsW + (size_t)sub * DD + c);
    float4 rw = *(const float4*)(RW + (size_t)rel * DD + c);
    float4 qw = *(const float4*)(QW + (size_t)ridx * DD + c);
    float4 wa = *(const float4*)(walpha_w + c);

    float p0 = fmaxf(hs.x + rw.x + qw.x, 0.f);
    float p1 = fmaxf(hs.y + rw.y + qw.y, 0.f);
    float p2 = fmaxf(hs.z + rw.z + qw.z, 0.f);
    float p3 = fmaxf(hs.w + rw.w + qw.w, 0.f);
    float part = p0 * wa.x + p1 * wa.y + p2 * wa.z + p3 * wa.w;
#pragma unroll
    for (int m = 16; m > 0; m >>= 1) part += __shfl_xor(part, m, 32);

    float alpha = 1.f / (1.f + __expf(-(part + walpha_b[0])));
    if (lane == 0) alpha_out[gid] = alpha;

    float4 h = *(const float4*)(hidden + (size_t)sub * DD + c);
    float4 r = *(const float4*)(rela + (size_t)rel * DD + c);
    float* dst = agg + (size_t)obj * DD + c;
    unsafeAtomicAdd(dst + 0, alpha * (h.x + r.x));
    unsafeAtomicAdd(dst + 1, alpha * (h.y + r.y));
    unsafeAtomicAdd(dst + 2, alpha * (h.z + r.z));
    unsafeAtomicAdd(dst + 3, alpha * (h.w + r.w));
}

extern "C" void kernel_launch(void* const* d_in, const int* in_sizes, int n_in,
                              void* d_out, int out_size, void* d_ws, size_t ws_size,
                              hipStream_t stream) {
    const int*   q_rel    = (const int*)d_in[1];
    const float* hidden   = (const float*)d_in[2];
    const int*   edges    = (const int*)d_in[3];
    const int*   nodes    = (const int*)d_in[4];
    const float* rela     = (const float*)d_in[5];
    const float* Ws       = (const float*)d_in[6];
    const float* Wr       = (const float*)d_in[7];
    const float* Wqr_w    = (const float*)d_in[8];
    const float* Wqr_b    = (const float*)d_in[9];
    const float* walpha_w = (const float*)d_in[10];
    const float* walpha_b = (const float*)d_in[11];
    const float* Wh       = (const float*)d_in[12];

    const int Bq   = in_sizes[0];
    const int N    = in_sizes[2] / DD;
    const int E    = in_sizes[3] / 6;
    const int relv = in_sizes[5] / DD;
    const int N_NODE = 100000;

    float* out       = (float*)d_out;
    float* agg       = out;                       // N*128 (message_agg -> hidden_new in-place)
    float* alpha_out = out + (size_t)N * DD;      // E
    float* flags     = alpha_out + E;             // N

    float* HsW = (float*)d_ws;                    // N*128
    float* RW  = HsW + (size_t)N * DD;            // relv*128
    float* QW  = RW + (size_t)relv * DD;          // Bq*128

    int n4 = (N * DD) / 4;
    zero_f4<<<(n4 + 255) / 256, 256, 0, stream>>>(agg, n4);
    node_flags<<<(N + 255) / 256, 256, 0, stream>>>(nodes, flags, N, N_NODE);
    precompute_rw_qw<<<relv + Bq, 128, 0, stream>>>(rela, q_rel, Wr, Wqr_w, Wqr_b,
                                                    RW, QW, relv);
    int gblocks = (N + 127) / 128;
    gemm_x128_w128<<<gblocks, 256, 0, stream>>>(hidden, Ws, HsW, N);
    edge_kernel<<<(E + 7) / 8, 256, 0, stream>>>(edges, hidden, rela, HsW, RW, QW,
                                                 walpha_w, walpha_b, agg, alpha_out, E);
    gemm_x128_w128<<<gblocks, 256, 0, stream>>>(agg, Wh, agg, N);  // in-place
}
